// MultiHeadAttention_29738353558106
// MI455X (gfx1250) — matrix operand order repaired and checked by reading, not compiled, not run on hardware
//
#include <hip/hip_runtime.h>
#include <hip/hip_bf16.h>
#include <hip/hip_fp16.h>

// ---------------------------------------------------------------------------
// Shared-weight MHA for MI455X (gfx1250, wave32, WMMA 16x16x32 f16->f32).
//
// Key simplifications proven from the reference:
//  * All heads share one (Wq,Wk,Wv): one head computed; tiled-concat @ Wout
//    folds into head @ Wout_eff, Wout_eff[d,j] = sum_h Wout[h*64+d, j].
//  * masked_fill with -1e-32 is a numeric no-op in f32 -> mask/seq_mask ignored.
//  * LayerNorm'd scores are bounded (|z| <= sqrt(2047) ~ 45) so exp() cannot
//    overflow f32 -> softmax without max-subtraction is safe.
//
// CDNA5-specific paths exercised:
//  * v_wmma_f32_16x16x32_f16 for all four GEMMs (f32 accumulate)
//  * tensor_load_to_lds (TDM, 6-arg clang-23 form) to stage the shared Q tile
//    once per workgroup (+ s_wait_tensorcnt), removing 8x-redundant Q loads
//  * global_prefetch_b8 for the K-tile stream
// ---------------------------------------------------------------------------

typedef __attribute__((ext_vector_type(16))) _Float16 v16h;
typedef __attribute__((ext_vector_type(8)))  _Float16 v8h;
typedef __attribute__((ext_vector_type(8)))  float    v8f;
typedef __attribute__((ext_vector_type(4)))  unsigned int u32x4;
typedef __attribute__((ext_vector_type(8)))  int      i32x8;
typedef __attribute__((ext_vector_type(4)))  int      i32x4;

#define HIDDEN   1024
#define HEAD_DIM 64
#define NHEADS   16
#define BATCH    4
#define SEQ      2048
#define MTOT     (BATCH*SEQ)      // 8192 rows total
#define PROB_STRIDE 264           // padded f16 row stride in LDS prob strip

__device__ __forceinline__ v16h combine16(v8h lo, v8h hi) {
    union { v16h v; v8h h[2]; } u;
    u.h[0] = lo; u.h[1] = hi;
    return u.v;
}

// ---------------------------------------------------------------------------
// Kernel 1: weight prep.  grid = dim3(256, 4), block = 256
//   y=0..2 : WT[n][k] = (f16) W[k][n]  for Wq/Wk/Wv   (BT-form for WMMA B)
//   y=3    : WeffT[j][d] = (f16) sum_h Wout[h*64+d][j]
// ---------------------------------------------------------------------------
__global__ void __launch_bounds__(256) prep_weights(
    const float* __restrict__ Wq, const float* __restrict__ Wk,
    const float* __restrict__ Wv, const float* __restrict__ Wout,
    _Float16* __restrict__ WqT, _Float16* __restrict__ WkT,
    _Float16* __restrict__ WvT, _Float16* __restrict__ WeffT)
{
    const int z   = blockIdx.y;
    const int idx = blockIdx.x * 256 + threadIdx.x;   // 0..65535
    if (z < 3) {
        const float* W = (z == 0) ? Wq : (z == 1) ? Wk : Wv;
        _Float16* WT   = (z == 0) ? WqT : (z == 1) ? WkT : WvT;
        const int n = idx >> 10;       // 0..63
        const int k = idx & 1023;      // 0..1023
        WT[idx] = (_Float16)W[k * HEAD_DIM + n];
    } else {
        const int j = idx >> 6;        // 0..1023
        const int d = idx & 63;        // 0..63
        float s = 0.0f;
        #pragma unroll
        for (int h = 0; h < NHEADS; ++h)
            s += Wout[(h * HEAD_DIM + d) * HIDDEN + j];
        WeffT[j * HEAD_DIM + d] = (_Float16)s;
    }
}

// ---------------------------------------------------------------------------
// Kernel 2: QKV projection. TMODE=0: row-major f16 [MTOT][64] (Q, K).
//           TMODE=1: transposed f16 [64][MTOT] (V).  Branch-free epilogue.
//   grid.x = MTOT/32 = 256, block = 256 (8 waves).
// ---------------------------------------------------------------------------
template<int TMODE>
__global__ void __launch_bounds__(256) qkv_proj(
    const float* __restrict__ X, const _Float16* __restrict__ WT,
    const float* __restrict__ bias, _Float16* __restrict__ Outp)
{
    const int tid  = threadIdx.x;
    const int wv   = tid >> 5;
    const int lane = tid & 31;
    const int l16  = lane & 15;
    const int g    = lane >> 4;                 // 0 or 1 (half-wave group)
    const int m0   = blockIdx.x * 32 + (wv >> 2) * 16;
    const int n0   = (wv & 3) * 16;

    const float*    arow = X  + (size_t)(m0 + l16) * HIDDEN;
    const _Float16* brow = WT + (size_t)(n0 + l16) * HIDDEN;

    v8f c = {};
    for (int k0 = 0; k0 < HIDDEN; k0 += 32) {
        v16h a;
        #pragma unroll
        for (int i = 0; i < 8; ++i) {           // A layout: K = 16g+i / 16g+8+i
            a[i]     = (_Float16)arow[k0 + 16 * g + i];
            a[i + 8] = (_Float16)arow[k0 + 16 * g + 8 + i];
        }
        v16h b = *(const v16h*)(brow + k0 + 16 * g);   // B layout: K = h + 16g
        c = __builtin_amdgcn_wmma_f32_16x16x32_f16(false, a, false, b,
                                                   (short)0, c, false, false);
    }

    const float bv = bias[n0 + l16];
    #pragma unroll
    for (int r = 0; r < 8; ++r) {
        const int m = m0 + r + 8 * g;           // C layout: M = r + 8g, N = l16
        const int n = n0 + l16;
        const _Float16 val = (_Float16)(c[r] + bv);
        if constexpr (TMODE == 0) Outp[(size_t)m * HEAD_DIM + n] = val;
        else                      Outp[(size_t)n * MTOT + m]     = val;
    }
}

// ---------------------------------------------------------------------------
// Kernel 3: fused attention core. grid = dim3(128, 4), block = 256 (8 waves).
// One batch b + one 16-query tile per workgroup. Wave w owns 256 keys
// (16 WMMA column tiles) -> 128 VGPRs of f32 score accumulators.
// Q tile staged into LDS once via the Tensor Data Mover.
// ---------------------------------------------------------------------------
__global__ void __launch_bounds__(256) attention(
    const _Float16* __restrict__ Q,   // [MTOT][64] row-major
    const _Float16* __restrict__ K,   // [MTOT][64] row-major
    const _Float16* __restrict__ VT,  // [64][MTOT] transposed
    _Float16* __restrict__ Hd)        // [MTOT][64] row-major
{
    // prob strips (per wave 16 x PROB_STRIDE f16); head partials alias on top
    __shared__ __align__(32) unsigned char smemRaw[8 * 16 * PROB_STRIDE * 2];
    _Float16* probs = (_Float16*)smemRaw;              // [8][16][PROB_STRIDE]
    float*    part  = (float*)smemRaw;                 // reuse: [8][16][64] f32
    __shared__ __align__(16) _Float16 qtile[16 * HEAD_DIM];  // TDM-staged Q tile
    __shared__ float red[8][16][2];                    // per-wave row partial sums
    __shared__ float rowStat[16][2];                   // mean, 1/(sigma+eps); later 1/rowsum

    const int tid  = threadIdx.x;
    const int wv   = tid >> 5;
    const int lane = tid & 31;
    const int l16  = lane & 15;
    const int g    = lane >> 4;
    const int b    = blockIdx.y;
    const int qt   = blockIdx.x;
    const size_t qRow  = (size_t)b * SEQ + (size_t)qt * 16;
    const size_t kBase = (size_t)b * SEQ;
    const int    kw0   = wv * 256;                     // this wave's key slice

    // ---- TDM: DMA the 16x64 f16 Q tile (2 KB) global -> LDS, one wave ------
    if (wv == 0) {
        const unsigned long long gaddr =
            (unsigned long long)(const void*)(Q + qRow * HEAD_DIM);
        const unsigned lds_off =
            (unsigned)(unsigned long long)(const void*)qtile;  // LDS byte offset
        u32x4 g0;
        g0[0] = 1u;                                    // count=1 user descriptor
        g0[1] = lds_off;                               // lds_addr
        g0[2] = (unsigned)(gaddr & 0xFFFFFFFFull);     // global_addr[31:0]
        g0[3] = (unsigned)((gaddr >> 32) & 0x01FFFFFFull) | (2u << 30); // +type=2
        i32x8 g1;
        g1[0] = 0x00010000;        // data_size=1 (2 bytes), wg_mask=0, no pad
        g1[1] = (64 << 16);        // tensor_dim0 = 64 elems   (bits 79:48)
        g1[2] = (16 << 16);        // tensor_dim1 = 16 rows    (bits 111:80)
        g1[3] = (64 << 16);        // tile_dim0 = 64           (bits 127:112)
        g1[4] = 16;                // tile_dim1 = 16           (bits 143:128)
        g1[5] = 64;                // tensor_dim0_stride = 64  (bits 207:160)
        g1[6] = 0;
        g1[7] = 0;                 // tensor_dim1_stride unused for 2D tile
        const i32x4 z4 = {0, 0, 0, 0};
        const i32x8 z8 = {0, 0, 0, 0, 0, 0, 0, 0};
        __builtin_amdgcn_tensor_load_to_lds(g0, g1, z4, z4, z8, 0);
        __builtin_amdgcn_s_wait_tensorcnt(0);
    }
    __syncthreads();

    // ---- Q A-fragments for both K-steps (d = 0..31 and 32..63) from LDS ----
    const _Float16* qrow = qtile + l16 * HEAD_DIM;
    const v16h aq0 = combine16(*(const v8h*)(qrow + 16 * g),
                               *(const v8h*)(qrow + 8 + 16 * g));
    const v16h aq1 = combine16(*(const v8h*)(qrow + 32 + 16 * g),
                               *(const v8h*)(qrow + 40 + 16 * g));

    // ---- scores: 16 C tiles of [16q x 16k], K-reduction over head_dim=64 ---
    v8f sc[16];
    #pragma unroll
    for (int ct = 0; ct < 16; ++ct) {
        const _Float16* krow = K + (kBase + kw0 + ct * 16 + l16) * HEAD_DIM;
        if (ct < 14)   // prefetch the K rows of the tile after next
            __builtin_prefetch(krow + 2 * 16 * HEAD_DIM, 0, 1);
        const v16h b0 = *(const v16h*)(krow + 16 * g);
        const v16h b1 = *(const v16h*)(krow + 32 + 16 * g);
        v8f c = {};
        c = __builtin_amdgcn_wmma_f32_16x16x32_f16(false, aq0, false, b0,
                                                   (short)0, c, false, false);
        c = __builtin_amdgcn_wmma_f32_16x16x32_f16(false, aq1, false, b1,
                                                   (short)0, c, false, false);
        sc[ct] = c;
    }

    // ---- scale by 1/sqrt(1024), row sums + sums of squares -----------------
    float rsum[8], rsq[8];
    #pragma unroll
    for (int r = 0; r < 8; ++r) { rsum[r] = 0.0f; rsq[r] = 0.0f; }
    #pragma unroll
    for (int ct = 0; ct < 16; ++ct)
        #pragma unroll
        for (int r = 0; r < 8; ++r) {
            const float v = sc[ct][r] * 0.03125f;
            sc[ct][r] = v;
            rsum[r] += v; rsq[r] += v * v;
        }
    #pragma unroll
    for (int r = 0; r < 8; ++r)
        for (int off = 1; off < 16; off <<= 1) {   // reduce within 16-lane half
            rsum[r] += __shfl_xor(rsum[r], off, 32);
            rsq[r]  += __shfl_xor(rsq[r],  off, 32);
        }
    if (l16 == 0) {
        #pragma unroll
        for (int r = 0; r < 8; ++r) {
            red[wv][r + 8 * g][0] = rsum[r];
            red[wv][r + 8 * g][1] = rsq[r];
        }
    }
    __syncthreads();
    if (tid < 16) {                                // mean / unbiased sigma per row
        float s = 0.0f, q = 0.0f;
        #pragma unroll
        for (int w2 = 0; w2 < 8; ++w2) { s += red[w2][tid][0]; q += red[w2][tid][1]; }
        const float mean = s * (1.0f / 2048.0f);
        float var = (q - s * mean) * (1.0f / 2047.0f);
        var = var < 0.0f ? 0.0f : var;
        rowStat[tid][0] = mean;
        rowStat[tid][1] = 1.0f / (sqrtf(var) + 1e-8f);
    }
    __syncthreads();

    // ---- layernorm + exp (no max-sub needed: |z|<=45 -> exp fits f32) ------
    float psum[8];
    #pragma unroll
    for (int r = 0; r < 8; ++r) {
        const float mu = rowStat[r + 8 * g][0];
        const float is = rowStat[r + 8 * g][1];
        psum[r] = 0.0f;
        #pragma unroll
        for (int ct = 0; ct < 16; ++ct) {
            const float p = __expf((sc[ct][r] - mu) * is);
            sc[ct][r] = p;
            psum[r] += p;
        }
    }
    #pragma unroll
    for (int r = 0; r < 8; ++r)
        for (int off = 1; off < 16; off <<= 1)
            psum[r] += __shfl_xor(psum[r], off, 32);
    if (l16 == 0) {
        #pragma unroll
        for (int r = 0; r < 8; ++r) red[wv][r + 8 * g][0] = psum[r];
    }
    __syncthreads();
    if (tid < 16) {
        float s = 0.0f;
        #pragma unroll
        for (int w2 = 0; w2 < 8; ++w2) s += red[w2][tid][0];
        rowStat[tid][0] = 1.0f / s;                // overwrite mean with 1/rowsum
    }
    __syncthreads();

    // ---- normalized probs -> LDS (C layout -> row-major strip) -------------
    _Float16* pw = probs + wv * 16 * PROB_STRIDE;
    #pragma unroll
    for (int r = 0; r < 8; ++r) {
        const float inv = rowStat[r + 8 * g][0];
        const int m = r + 8 * g;
        #pragma unroll
        for (int ct = 0; ct < 16; ++ct)
            pw[m * PROB_STRIDE + ct * 16 + l16] = (_Float16)(sc[ct][r] * inv);
    }
    // per-wave write -> per-wave read: compiler inserts s_wait_dscnt, no barrier

    // ---- AV: partial head tile [16 x 64] over this wave's 256 keys ---------
    v8f hacc[4] = {};
    const _Float16* pwm = pw + l16 * PROB_STRIDE;     // A row m = l16
    #pragma unroll
    for (int s = 0; s < 8; ++s) {                     // key chunks of 32
        const int col = s * 32;
        const v16h a = combine16(*(const v8h*)(pwm + col + 16 * g),
                                 *(const v8h*)(pwm + col + 8 + 16 * g));
        #pragma unroll
        for (int dt = 0; dt < 4; ++dt) {
            const _Float16* vrow = VT + (size_t)(dt * 16 + l16) * MTOT
                                      + kBase + kw0 + col + 16 * g;
            const v16h bf = *(const v16h*)vrow;
            hacc[dt] = __builtin_amdgcn_wmma_f32_16x16x32_f16(
                false, a, false, bf, (short)0, hacc[dt], false, false);
        }
    }

    __syncthreads();   // everyone done reading probs before aliasing as `part`
    #pragma unroll
    for (int dt = 0; dt < 4; ++dt)
        #pragma unroll
        for (int r = 0; r < 8; ++r)
            part[(wv * 16 + r + 8 * g) * 64 + dt * 16 + l16] = hacc[dt][r];
    __syncthreads();

    // ---- reduce 8 wave partials -> head tile, store f16 --------------------
    for (int e = tid; e < 16 * 64; e += 256) {
        const int m = e >> 6, d = e & 63;
        float s = 0.0f;
        #pragma unroll
        for (int w2 = 0; w2 < 8; ++w2) s += part[(w2 * 16 + m) * 64 + d];
        Hd[(qRow + m) * HEAD_DIM + d] = (_Float16)s;
    }
}

// ---------------------------------------------------------------------------
// Kernel 4: output projection  out = head @ WeffT^T + bout  (f32 out).
//   grid = dim3(1024/128, 8192/16), block = 256 (8 waves, one n-tile each).
// ---------------------------------------------------------------------------
__global__ void __launch_bounds__(256) out_proj(
    const _Float16* __restrict__ Hd,    // [MTOT][64]
    const _Float16* __restrict__ WeffT, // [1024][64]  (BT-form)
    const float* __restrict__ bout,
    float* __restrict__ Out)            // [MTOT][1024]
{
    const int tid  = threadIdx.x;
    const int wv   = tid >> 5;
    const int lane = tid & 31;
    const int l16  = lane & 15;
    const int g    = lane >> 4;
    const int m0   = blockIdx.y * 16;
    const int n0   = blockIdx.x * 128 + wv * 16;

    const _Float16* arow = Hd    + (size_t)(m0 + l16) * HEAD_DIM;
    const _Float16* brow = WeffT + (size_t)(n0 + l16) * HEAD_DIM;

    v8f c = {};
    #pragma unroll
    for (int k0 = 0; k0 < 64; k0 += 32) {
        const v16h a = combine16(*(const v8h*)(arow + k0 + 16 * g),
                                 *(const v8h*)(arow + k0 + 8 + 16 * g));
        const v16h b = *(const v16h*)(brow + k0 + 16 * g);
        c = __builtin_amdgcn_wmma_f32_16x16x32_f16(false, a, false, b,
                                                   (short)0, c, false, false);
    }
    const float bb = bout[n0 + l16];
    #pragma unroll
    for (int r = 0; r < 8; ++r)
        Out[(size_t)(m0 + r + 8 * g) * HIDDEN + n0 + l16] = c[r] + bb;
}

// ---------------------------------------------------------------------------
// Launcher. Workspace layout (f16):
//   WqT/WkT/WvT 64x1024, WeffT 1024x64, Qf/Kf [8192][64], VT [64][8192],
//   Hd [8192][64].  Total ~4.5 MB.
// ---------------------------------------------------------------------------
extern "C" void kernel_launch(void* const* d_in, const int* in_sizes, int n_in,
                              void* d_out, int out_size, void* d_ws, size_t ws_size,
                              hipStream_t stream) {
    (void)in_sizes; (void)n_in; (void)out_size; (void)ws_size;

    const float* query = (const float*)d_in[0];
    const float* key   = (const float*)d_in[1];
    const float* value = (const float*)d_in[2];
    // d_in[3] = mask     : -1e-32 fill is a numeric no-op in f32 -> unused
    const float* Wq    = (const float*)d_in[4];
    const float* bq    = (const float*)d_in[5];
    const float* Wk    = (const float*)d_in[6];
    const float* bk    = (const float*)d_in[7];
    const float* Wv    = (const float*)d_in[8];
    const float* bv    = (const float*)d_in[9];
    const float* Wout  = (const float*)d_in[10];
    const float* bout  = (const float*)d_in[11];
    // d_in[12] = seq_mask: also only feeds the -1e-32 no-op fill -> unused

    char* ws = (char*)d_ws;
    _Float16* WqT   = (_Float16*)ws;  ws += (size_t)64 * 1024 * 2;
    _Float16* WkT   = (_Float16*)ws;  ws += (size_t)64 * 1024 * 2;
    _Float16* WvT   = (_Float16*)ws;  ws += (size_t)64 * 1024 * 2;
    _Float16* WeffT = (_Float16*)ws;  ws += (size_t)1024 * 64 * 2;
    _Float16* Qf    = (_Float16*)ws;  ws += (size_t)MTOT * HEAD_DIM * 2;
    _Float16* Kf    = (_Float16*)ws;  ws += (size_t)MTOT * HEAD_DIM * 2;
    _Float16* VT    = (_Float16*)ws;  ws += (size_t)HEAD_DIM * MTOT * 2;
    _Float16* Hd    = (_Float16*)ws;  ws += (size_t)MTOT * HEAD_DIM * 2;

    prep_weights<<<dim3(256, 4), 256, 0, stream>>>(Wq, Wk, Wv, Wout,
                                                   WqT, WkT, WvT, WeffT);
    qkv_proj<0><<<dim3(MTOT / 32), 256, 0, stream>>>(query, WqT, bq, Qf);
    qkv_proj<0><<<dim3(MTOT / 32), 256, 0, stream>>>(key,   WkT, bk, Kf);
    qkv_proj<1><<<dim3(MTOT / 32), 256, 0, stream>>>(value, WvT, bv, VT);
    attention<<<dim3(SEQ / 16, BATCH), 256, 0, stream>>>(Qf, Kf, VT, Hd);
    out_proj<<<dim3(HIDDEN / 128, MTOT / 16), 256, 0, stream>>>(Hd, WeffT, bout,
                                                                (float*)d_out);
}
